// PNAModel_51479478010220
// MI455X (gfx1250) — compile-verified, matrix operand order checked by reading, not run on hardware
//
#include <hip/hip_runtime.h>

#define Fc 64
#define Lc 4
#define DPOST 832              // (4*3+1)*64
#define EPSV 1e-5f

typedef __attribute__((ext_vector_type(16))) __bf16 v16bf;
typedef __attribute__((ext_vector_type(8)))  __bf16 v8bf;
typedef __attribute__((ext_vector_type(8)))  float  v8f;

// ---- helpers -------------------------------------------------------------
static __device__ __forceinline__ unsigned short f2bfu(float f) {
  unsigned u = __float_as_uint(f);
  u += 0x7FFFu + ((u >> 16) & 1u);          // round-to-nearest-even to bf16
  return (unsigned short)(u >> 16);
}
static __device__ __forceinline__ float bfu2f(unsigned short s) {
  return __uint_as_float((unsigned)s << 16);
}
// build a v16bf fragment from two contiguous 16-byte chunks (t=0..7 at p0, t=8..15 at p0+16)
static __device__ __forceinline__ v16bf ldfrag(const unsigned short* p0) {
  v8bf lo = __builtin_bit_cast(v8bf, *(const uint4*)(p0));
  v8bf hi = __builtin_bit_cast(v8bf, *(const uint4*)(p0 + 16));
  return __builtin_shufflevector(lo, hi, 0,1,2,3,4,5,6,7,8,9,10,11,12,13,14,15);
}
// order-preserving float<->uint encoding so atomicMin/Max on uint == float min/max
static __device__ __forceinline__ unsigned encf(float f) {
  unsigned u = __float_as_uint(f);
  return (u & 0x80000000u) ? ~u : (u | 0x80000000u);
}
static __device__ __forceinline__ float decf(unsigned u) {
  unsigned b = (u & 0x80000000u) ? (u & 0x7FFFFFFFu) : ~u;
  return __uint_as_float(b);
}

// ---- small setup kernels -------------------------------------------------
__global__ void k_zero_f(float* p, long n) {
  long i = (long)blockIdx.x * 256 + threadIdx.x;
  if (i < n) p[i] = 0.f;
}

__global__ void k_count(const int* __restrict__ ei, float* __restrict__ count, long E) {
  long e = (long)blockIdx.x * 256 + threadIdx.x;
  if (e < E) atomicAdd(&count[ei[E + e]], 1.0f);
}

__global__ void k_hist(const float* __restrict__ dh, int n, float* __restrict__ avg) {
  __shared__ float sa[256], sb[256];
  float a = 0.f, b = 0.f;
  for (int i = threadIdx.x; i < n; i += 256) {
    float h = dh[i];
    a += logf((float)i + 1.0f) * h;
    b += h;
  }
  sa[threadIdx.x] = a; sb[threadIdx.x] = b;
  __syncthreads();
  for (int off = 128; off > 0; off >>= 1) {
    if ((int)threadIdx.x < off) { sa[threadIdx.x] += sa[threadIdx.x + off]; sb[threadIdx.x] += sb[threadIdx.x + off]; }
    __syncthreads();
  }
  if (threadIdx.x == 0) avg[0] = sa[0] / sb[0];
}

__global__ void k_scalers(const float* __restrict__ count, const float* __restrict__ avg,
                          float* __restrict__ amp, float* __restrict__ att, long N) {
  long i = (long)blockIdx.x * 256 + threadIdx.x;
  if (i < N) {
    float d  = fmaxf(count[i], 1.f);
    float lg = logf(d + 1.f);
    float al = avg[0];
    amp[i] = lg / al;
    att[i] = al / lg;
  }
}

// plain fp32 -> bf16 convert (for x -> h_bf16)
__global__ void k_cvt_bf(const float* __restrict__ in, unsigned short* __restrict__ out, long n) {
  long i = (long)blockIdx.x * 256 + threadIdx.x;
  if (i < n) out[i] = f2bfu(in[i]);
}

// transposing fp32 -> bf16 convert: in [L][K][64] -> out [L][64][K]
// makes every WMMA B-fragment two contiguous b128 loads
__global__ void k_cvt_bf_t(const float* __restrict__ in, unsigned short* __restrict__ out,
                           int K, long total) {
  long idx = (long)blockIdx.x * 256 + threadIdx.x;
  if (idx >= total) return;
  int  n = (int)(idx & 63);
  long r = idx >> 6;            // = l*K + k
  long k = r % K;
  long l = r / K;
  out[(l * 64 + n) * (long)K + k] = f2bfu(in[idx]);
}

__global__ void k_initagg(float* __restrict__ s1, float* __restrict__ s2,
                          unsigned* __restrict__ mn, unsigned* __restrict__ mx, long n) {
  long i = (long)blockIdx.x * 256 + threadIdx.x;
  if (i < n) { s1[i] = 0.f; s2[i] = 0.f; mn[i] = 0xFFFFFFFFu; mx[i] = 0u; }
}

// ---- edge message GEMM + scatter aggregation -----------------------------
// One wave32 per 16-edge tile. A = [16 x 128] bf16 (concat h[dst]|h[src]) staged
// in LDS via uint4 copies; B = W_pre^T bf16 (L2 resident) as b128 fragment loads.
// 4 K-steps x 4 N-tiles = 16 v_wmma_f32_16x16x32_bf16, then atomic scatter into
// L2-resident aggregation buffers.
__global__ void __launch_bounds__(128) k_msg(
    const unsigned short* __restrict__ hbf, const int* __restrict__ ei,
    const unsigned short* __restrict__ wpreT, const float* __restrict__ bpre,
    float* __restrict__ s1, float* __restrict__ s2,
    unsigned* __restrict__ mn, unsigned* __restrict__ mx,
    long N, long E, int layer)
{
  __shared__ unsigned short Abuf[4][16][128];   // 4 waves * 4KB
  __shared__ int dstbuf[4][16];
  const int wave = threadIdx.x >> 5;
  const int lane = threadIdx.x & 31;
  const long ebase = ((long)blockIdx.x * 4 + wave) * 16;

  { // gather: 2 lanes per edge row; half 0 -> h[dst] cols 0..63, half 1 -> h[src] cols 64..127
    int row = lane >> 1, half = lane & 1;
    long e = ebase + row; if (e >= E) e = E - 1;
    int sn = ei[e], dn = ei[E + e];
    if (half == 0) dstbuf[wave][row] = dn;
    const uint4* hr = (const uint4*)(hbf + (long)(half ? sn : dn) * Fc);  // 64 bf16 = 8 x 16B
    uint4* dp = (uint4*)&Abuf[wave][row][half * 64];
#pragma unroll
    for (int j = 0; j < 8; ++j) dp[j] = hr[j];
  }
  __syncthreads();

  const int nlo = lane & 15;
  const int hi8 = (lane >> 4) * 8;                 // K offset split per 16-bit A/B layout
  const unsigned short* WT = wpreT + (long)layer * 64 * 128;   // [64][128] n-major

  v16bf afr[4];
#pragma unroll
  for (int kb = 0; kb < 4; ++kb)
    afr[kb] = ldfrag(&Abuf[wave][nlo][kb * 32 + hi8]);

#pragma unroll
  for (int nt = 0; nt < 4; ++nt) {
    v8f acc = {};
#pragma unroll
    for (int kb = 0; kb < 4; ++kb) {
      v16bf b = ldfrag(WT + (long)(nt * 16 + nlo) * 128 + kb * 32 + hi8);
      acc = __builtin_amdgcn_wmma_f32_16x16x32_bf16(false, afr[kb], false, b,
                                                    (short)0, acc, false, false);
    }
    const int ch = nt * 16 + nlo;
    const float bias = bpre[layer * Fc + ch];
#pragma unroll
    for (int r = 0; r < 8; ++r) {
      int M = r + hi8;
      long e2 = ebase + M;
      if (e2 < E) {
        float v = acc[r] + bias;
        long off = (long)dstbuf[wave][M] * Fc + ch;
        atomicAdd(&s1[off], v);
        atomicAdd(&s2[off], v * v);
        unsigned ev = encf(v);
        atomicMax(&mx[off], ev);
        atomicMin(&mn[off], ev);
      }
    }
  }
}

// ---- finalize aggregators -> bf16 concat row [h | agg | agg*amp | agg*att] ----
__global__ void k_xcat(const unsigned short* __restrict__ hbf, const float* __restrict__ count,
                       const float* __restrict__ amp, const float* __restrict__ att,
                       const float* __restrict__ s1, const float* __restrict__ s2,
                       const unsigned* __restrict__ mn, const unsigned* __restrict__ mx,
                       unsigned short* __restrict__ xcat, long N)
{
  long idx = (long)blockIdx.x * 256 + threadIdx.x;
  if (idx >= N * (long)Fc) return;
  long i = idx >> 6; int c = (int)(idx & 63);
  float cv   = count[i];
  float cnt  = fmaxf(cv, 1.f);
  float mean = s1[idx] / cnt;
  float var  = s2[idx] / cnt - mean * mean;
  float sd   = sqrtf(fmaxf(var, 0.f) + EPSV);
  float mnv  = (cv > 0.f) ? decf(mn[idx]) : 0.f;
  float mxv  = (cv > 0.f) ? decf(mx[idx]) : 0.f;
  unsigned short* row = xcat + i * DPOST;
  row[c] = hbf[idx];
  float a4[4] = { mean, mnv, mxv, sd };
  float am = amp[i], at = att[i];
#pragma unroll
  for (int j = 0; j < 4; ++j) {
    row[ 64 + j * 64 + c] = f2bfu(a4[j]);
    row[320 + j * 64 + c] = f2bfu(a4[j] * am);
    row[576 + j * 64 + c] = f2bfu(a4[j] * at);
  }
}

// ---- post_nn GEMM ([N,832]@[832,64]) then lin GEMM ([N,64]@[64,64]) -------
__global__ void __launch_bounds__(128) k_post(
    const unsigned short* __restrict__ xcat,
    const unsigned short* __restrict__ wpostT, const float* __restrict__ bpost,
    const unsigned short* __restrict__ wlinT,  const float* __restrict__ blin,
    unsigned short* __restrict__ houtbf, float* __restrict__ outf,
    long N, int layer, int relu)
{
  __shared__ unsigned short out1[4][16][64];     // per-wave 16x64 intermediate (bf16)
  const int wave = threadIdx.x >> 5, lane = threadIdx.x & 31;
  const int nlo = lane & 15, hi8 = (lane >> 4) * 8;
  const long nbase = ((long)blockIdx.x * 4 + wave) * 16;
  long arow = nbase + nlo; if (arow >= N) arow = N - 1;
  const unsigned short* xr = xcat + arow * DPOST;
  const unsigned short* Wp = wpostT + (long)layer * 64 * DPOST;   // [64][832] n-major

  v8f acc[4]; { v8f z = {}; acc[0] = z; acc[1] = z; acc[2] = z; acc[3] = z; }
  for (int kb = 0; kb < 26; ++kb) {              // 26 * 32 = 832 K
    v16bf a = ldfrag(xr + kb * 32 + hi8);
#pragma unroll
    for (int nt = 0; nt < 4; ++nt) {
      v16bf b = ldfrag(Wp + (long)(nt * 16 + nlo) * DPOST + kb * 32 + hi8);
      acc[nt] = __builtin_amdgcn_wmma_f32_16x16x32_bf16(false, a, false, b,
                                                        (short)0, acc[nt], false, false);
    }
  }
#pragma unroll
  for (int nt = 0; nt < 4; ++nt) {               // + b_post, re-fragment via LDS
    const int ch = nt * 16 + nlo;
    const float bias = bpost[layer * Fc + ch];
#pragma unroll
    for (int r = 0; r < 8; ++r) out1[wave][r + hi8][ch] = f2bfu(acc[nt][r] + bias);
  }
  __syncthreads();

  const unsigned short* Wl = wlinT + (long)layer * 64 * 64;       // [64][64] n-major
  v16bf a2[2];
#pragma unroll
  for (int kb = 0; kb < 2; ++kb)
    a2[kb] = ldfrag(&out1[wave][nlo][kb * 32 + hi8]);

#pragma unroll
  for (int nt = 0; nt < 4; ++nt) {
    v8f c = {};
#pragma unroll
    for (int kb = 0; kb < 2; ++kb) {
      v16bf b = ldfrag(Wl + (long)(nt * 16 + nlo) * 64 + kb * 32 + hi8);
      c = __builtin_amdgcn_wmma_f32_16x16x32_bf16(false, a2[kb], false, b,
                                                  (short)0, c, false, false);
    }
    const int ch = nt * 16 + nlo;
    const float bias = blin[layer * Fc + ch];
#pragma unroll
    for (int r = 0; r < 8; ++r) {
      long nr = nbase + r + hi8;
      if (nr < N) {
        float v = c[r] + bias;
        if (relu) v = fmaxf(v, 0.f);
        houtbf[nr * Fc + ch] = f2bfu(v);         // bf16 activations for next layer
        if (outf) outf[nr * Fc + ch] = v;        // fp32 result (final layer only)
      }
    }
  }
}

// ---- host orchestration ---------------------------------------------------
extern "C" void kernel_launch(void* const* d_in, const int* in_sizes, int n_in,
                              void* d_out, int out_size, void* d_ws, size_t ws_size,
                              hipStream_t stream)
{
  (void)n_in; (void)out_size; (void)ws_size;
  const float* x     = (const float*)d_in[0];
  const int*   ei    = (const int*)d_in[1];
  const float* dh    = (const float*)d_in[2];
  const float* Wpre  = (const float*)d_in[3];
  const float* bpre  = (const float*)d_in[4];
  const float* Wpost = (const float*)d_in[5];
  const float* bpost = (const float*)d_in[6];
  const float* Wlin  = (const float*)d_in[7];
  const float* blin  = (const float*)d_in[8];
  float* out = (float*)d_out;

  const long N  = in_sizes[0] / Fc;
  const long E  = in_sizes[1] / 2;
  const int  DH = in_sizes[2];

  char* ws = (char*)d_ws;
  size_t off = 0;
  auto alloc = [&](size_t bytes) -> char* {
    char* p = ws + off;
    off = (off + bytes + 255) & ~(size_t)255;
    return p;
  };
  float*          count = (float*)alloc((size_t)N * 4);
  float*          amp   = (float*)alloc((size_t)N * 4);
  float*          att   = (float*)alloc((size_t)N * 4);
  float*          avg   = (float*)alloc(256);
  float*          s1    = (float*)alloc((size_t)N * Fc * 4);
  float*          s2    = (float*)alloc((size_t)N * Fc * 4);
  unsigned*       mn    = (unsigned*)alloc((size_t)N * Fc * 4);
  unsigned*       mx    = (unsigned*)alloc((size_t)N * Fc * 4);
  unsigned short* xcat  = (unsigned short*)alloc((size_t)N * DPOST * 2);
  unsigned short* wpreT = (unsigned short*)alloc((size_t)Lc * 128 * 64 * 2);
  unsigned short* wpostT= (unsigned short*)alloc((size_t)Lc * DPOST * 64 * 2);
  unsigned short* wlinT = (unsigned short*)alloc((size_t)Lc * 64 * 64 * 2);
  unsigned short* hbf0  = (unsigned short*)alloc((size_t)N * Fc * 2);
  unsigned short* hbf1  = (unsigned short*)alloc((size_t)N * Fc * 2);

  dim3 b256(256);
  k_zero_f<<<dim3((unsigned)((N + 255) / 256)), b256, 0, stream>>>(count, N);
  k_count<<<dim3((unsigned)((E + 255) / 256)), b256, 0, stream>>>(ei, count, E);
  k_hist<<<dim3(1), b256, 0, stream>>>(dh, DH, avg);
  k_scalers<<<dim3((unsigned)((N + 255) / 256)), b256, 0, stream>>>(count, avg, amp, att, N);

  const long nwpre  = (long)Lc * 128 * 64;
  const long nwpost = (long)Lc * DPOST * 64;
  const long nwlin  = (long)Lc * 64 * 64;
  k_cvt_bf_t<<<dim3((unsigned)((nwpre  + 255) / 256)), b256, 0, stream>>>(Wpre,  wpreT,  128,  nwpre);
  k_cvt_bf_t<<<dim3((unsigned)((nwpost + 255) / 256)), b256, 0, stream>>>(Wpost, wpostT, DPOST, nwpost);
  k_cvt_bf_t<<<dim3((unsigned)((nwlin  + 255) / 256)), b256, 0, stream>>>(Wlin,  wlinT,  64,   nwlin);

  const long NF = N * Fc;
  k_cvt_bf<<<dim3((unsigned)((NF + 255) / 256)), b256, 0, stream>>>(x, hbf0, NF);

  const unsigned etiles = (unsigned)((E + 15) / 16);
  const unsigned ntiles = (unsigned)((N + 15) / 16);
  const unsigned short* hcur = hbf0;
  for (int l = 0; l < Lc; ++l) {
    k_initagg<<<dim3((unsigned)((NF + 255) / 256)), b256, 0, stream>>>(s1, s2, mn, mx, NF);
    k_msg<<<dim3((etiles + 3) / 4), dim3(128), 0, stream>>>(hcur, ei, wpreT, bpre,
                                                            s1, s2, mn, mx, N, E, l);
    k_xcat<<<dim3((unsigned)((NF + 255) / 256)), b256, 0, stream>>>(hcur, count, amp, att,
                                                                    s1, s2, mn, mx, xcat, N);
    unsigned short* hnext = (l & 1) ? hbf0 : hbf1;
    float* outf = (l == Lc - 1) ? out : nullptr;
    k_post<<<dim3((ntiles + 3) / 4), dim3(128), 0, stream>>>(xcat, wpostT, bpost, wlinT, blin,
                                                             hnext, outf, N, l, (l < Lc - 1) ? 1 : 0);
    hcur = hnext;
  }
}
